// ScaleDotProductAttention_28870770163962
// MI455X (gfx1250) — compile-verified, matrix operand order checked by reading, not run on hardware
//
#include <hip/hip_runtime.h>

// ---- CDNA5 WMMA types -------------------------------------------------------
typedef __attribute__((ext_vector_type(16))) __bf16 v16bf;
typedef __attribute__((ext_vector_type(8)))  __bf16 v8bf;
typedef __attribute__((ext_vector_type(8)))  float  v8f;
typedef __attribute__((ext_vector_type(4)))  float  v4f;

union V16 { v16bf v; struct { v8bf lo; v8bf hi; } p; };

// Problem shape (fixed by reference setup_inputs)
constexpr int N_B   = 8;
constexpr int L     = 2048;
constexpr int DM    = 1024;   // d_model
constexpr int DK    = 128;    // d_k

// LDS K-tile row stride: 128 + 8 pad elements -> 272B rows, avoids the
// 256B (=64-bank) period so 16-lane ds_load_b128 fragments hit all banks.
constexpr int KSTRIDE = 136;

__device__ __forceinline__ void split_bf16(float f, __bf16& hi, __bf16& lo) {
    hi = (__bf16)f;
    lo = (__bf16)(f - (float)hi);
}

__device__ __forceinline__ v8f wmma3(const v16bf& ah, const v16bf& al,
                                     const v16bf& bh, const v16bf& bl, v8f c) {
    // bf16 x 3 emulation of an fp32 MAC: a*b ~= ah*bh + al*bh + ah*bl
    c = __builtin_amdgcn_wmma_f32_16x16x32_bf16(false, ah, false, bh, (short)0, c, false, false);
    c = __builtin_amdgcn_wmma_f32_16x16x32_bf16(false, al, false, bh, (short)0, c, false, false);
    c = __builtin_amdgcn_wmma_f32_16x16x32_bf16(false, ah, false, bl, (short)0, c, false, false);
    return c;
}

// LDS byte-offset of a pointer into a __shared__ array (AS3 ptrtoint).
#define LDS_OFF(p) ((unsigned)(uintptr_t)(__attribute__((address_space(3))) const __bf16*)(p))

// ---------------------------------------------------------------------------
// Phase 1: Q = x@WQ, K = x@WK, stored to workspace as bf16 (hi,lo) pairs.
// Block = 256 threads = 8 waves. Waves 0-3 -> Q, waves 4-7 -> K.
// Each wave computes a 16x128 strip (8 WMMA tiles), K-loop over d_model.
// ---------------------------------------------------------------------------
__global__ __launch_bounds__(256)
void proj_qk_kernel(const float* __restrict__ x,
                    const float* __restrict__ WQ,
                    const float* __restrict__ WK,
                    __bf16* __restrict__ qh, __bf16* __restrict__ ql,
                    __bf16* __restrict__ kh, __bf16* __restrict__ kl) {
    const int n    = blockIdx.y;
    const int tid  = threadIdx.x;
    const int wave = tid >> 5;
    const int lane = tid & 31;
    const int h    = lane >> 4;   // half-wave (selects K sub-range)
    const int m    = lane & 15;   // row within A tile / col within B tile

    const bool isK = (wave >= 4);
    const int  rt  = wave & 3;
    const int  row = blockIdx.x * 64 + rt * 16 + m;

    const float* __restrict__ W    = isK ? WK : WQ;
    __bf16* __restrict__      outh = isK ? kh : qh;
    __bf16* __restrict__      outl = isK ? kl : ql;

    const float* xrow = x + ((size_t)n * L + row) * DM;

    v8f acc[8] = {};

    for (int k0 = 0; k0 < DM; k0 += 32) {
        // --- A fragment: x rows, contiguous 8-float chunks per half-wave ---
        v16bf ah, al;
        {
            const float* p0 = xrow + k0 + 8 * h;        // K = k0+8h .. +7
            const float* p1 = xrow + k0 + 16 + 8 * h;   // K = k0+16+8h .. +7
#pragma unroll
            for (int e = 0; e < 8; ++e) {
                __bf16 hi, lo;
                split_bf16(p0[e], hi, lo); ah[e] = hi;     al[e] = lo;
                split_bf16(p1[e], hi, lo); ah[8 + e] = hi; al[8 + e] = lo;
            }
        }
#pragma unroll
        for (int t = 0; t < 8; ++t) {
            const int col = t * 16 + m;                 // B-operand lane column
            v16bf bh, bl;
#pragma unroll
            for (int e = 0; e < 8; ++e) {
                __bf16 hi, lo;
                split_bf16(W[(k0 + 8 * h + e) * DK + col], hi, lo);
                bh[e] = hi; bl[e] = lo;
                split_bf16(W[(k0 + 16 + 8 * h + e) * DK + col], hi, lo);
                bh[8 + e] = hi; bl[8 + e] = lo;
            }
            acc[t] = wmma3(ah, al, bh, bl, acc[t]);
        }
    }

    // Store 16x128 strip as bf16 hi/lo. C layout: VGPR r -> row 8h+r, col = m.
    const int rowbase = blockIdx.x * 64 + rt * 16 + 8 * h;
#pragma unroll
    for (int t = 0; t < 8; ++t) {
        const int col = t * 16 + m;
#pragma unroll
        for (int r = 0; r < 8; ++r) {
            __bf16 hi, lo;
            split_bf16(acc[t][r], hi, lo);
            const size_t idx = ((size_t)n * L + rowbase + r) * DK + col;
            outh[idx] = hi;
            outl[idx] = lo;
        }
    }
}

// ---------------------------------------------------------------------------
// Phase 2: scores = (Q @ K^T) / sqrt(d_k), masked (pad + causal).
// One 128x128 output tile per block; 8 waves, each owns a 16x128 strip.
// K-tile (B operands) staged once per block into LDS with async-to-LDS
// loads (ASYNCcnt path); fully-future tiles (bcol > brow) fill -inf.
// ---------------------------------------------------------------------------
__global__ __launch_bounds__(256)
void scores_kernel(const __bf16* __restrict__ qh, const __bf16* __restrict__ ql,
                   const __bf16* __restrict__ kh, const __bf16* __restrict__ kl,
                   const float* __restrict__ masked_info,
                   float* __restrict__ out) {
    __shared__ __bf16 lds_kh[128 * KSTRIDE];
    __shared__ __bf16 lds_kl[128 * KSTRIDE];

    const int n    = blockIdx.z;
    const int brow = blockIdx.y;
    const int bcol = blockIdx.x;
    const int tid  = threadIdx.x;

    float* outn = out + (size_t)n * L * L;

    if (bcol > brow) {
        // Entire tile strictly above the diagonal -> -inf everywhere.
        // (Block-uniform branch: no barrier divergence.)
        const float ninf = -__builtin_inff();
        const v4f fill = { ninf, ninf, ninf, ninf };
        for (int i = tid; i < (128 * 128) / 4; i += 256) {
            const int r  = i >> 5;        // 32 x float4 per row
            const int c4 = i & 31;
            v4f* p = (v4f*)(outn + (size_t)(brow * 128 + r) * L + bcol * 128);
            p[c4] = fill;
        }
        return;
    }

    // ---- Stage the 128x128 bf16 hi/lo K-tile into LDS (async, 16B/lane) ----
    {
        const __bf16* src_h = kh + ((size_t)n * L + bcol * 128) * DK;
        const __bf16* src_l = kl + ((size_t)n * L + bcol * 128) * DK;
        const unsigned base_h = LDS_OFF(lds_kh);
        const unsigned base_l = LDS_OFF(lds_kl);
#pragma unroll
        for (int i = tid; i < (128 * DK) / 8; i += 256) {   // 2048 x 16B chunks
            const int r = i >> 4;          // K-tile row
            const int c = i & 15;          // 16B chunk within row
            const __bf16* gh = src_h + r * DK + c * 8;
            const __bf16* gl = src_l + r * DK + c * 8;
            const unsigned lh = base_h + (unsigned)((r * KSTRIDE + c * 8) * 2);
            const unsigned ll = base_l + (unsigned)((r * KSTRIDE + c * 8) * 2);
            asm volatile("global_load_async_to_lds_b128 %0, %1, off"
                         :: "v"(lh), "v"(gh) : "memory");
            asm volatile("global_load_async_to_lds_b128 %0, %1, off"
                         :: "v"(ll), "v"(gl) : "memory");
        }
        asm volatile("s_wait_asynccnt 0x0" ::: "memory");
    }
    __syncthreads();

    const int wave = tid >> 5;
    const int lane = tid & 31;
    const int h    = lane >> 4;
    const int m    = lane & 15;

    const int arow = brow * 128 + wave * 16 + m;
    const __bf16* qhr = qh + ((size_t)n * L + arow) * DK;
    const __bf16* qlr = ql + ((size_t)n * L + arow) * DK;

    v8f acc[8] = {};

    for (int k0 = 0; k0 < DK; k0 += 32) {
        V16 Ah, Al;
        Ah.p.lo = *(const v8bf*)(qhr + k0 + 8 * h);
        Ah.p.hi = *(const v8bf*)(qhr + k0 + 16 + 8 * h);
        Al.p.lo = *(const v8bf*)(qlr + k0 + 8 * h);
        Al.p.hi = *(const v8bf*)(qlr + k0 + 16 + 8 * h);
#pragma unroll
        for (int t = 0; t < 8; ++t) {
            // B operand = K^T: lane column m picks K-matrix row (from LDS).
            const __bf16* khr = &lds_kh[(t * 16 + m) * KSTRIDE];
            const __bf16* klr = &lds_kl[(t * 16 + m) * KSTRIDE];
            V16 Bh, Bl;
            Bh.p.lo = *(const v8bf*)(khr + k0 + 8 * h);
            Bh.p.hi = *(const v8bf*)(khr + k0 + 16 + 8 * h);
            Bl.p.lo = *(const v8bf*)(klr + k0 + 8 * h);
            Bl.p.hi = *(const v8bf*)(klr + k0 + 16 + 8 * h);
            acc[t] = wmma3(Ah.v, Al.v, Bh.v, Bl.v, acc[t]);
        }
    }

    // Epilogue: scale + pad mask + causal mask, then store.
    const float inv_scale = 0.08838834764831845f;  // 1/sqrt(128)
    const float ninf = -__builtin_inff();
    const float* mi = masked_info + (size_t)n * L;

    const int row0 = brow * 128 + wave * 16 + 8 * h;  // + r
    float mrow[8];
#pragma unroll
    for (int r = 0; r < 8; ++r) mrow[r] = mi[row0 + r];

#pragma unroll
    for (int t = 0; t < 8; ++t) {
        const int col = bcol * 128 + t * 16 + m;
        const float mcol = mi[col];
#pragma unroll
        for (int r = 0; r < 8; ++r) {
            const int row = row0 + r;
            const bool msk = (col > row) || (mrow[r] * mcol == 0.0f);
            outn[(size_t)row * L + col] = msk ? ninf : acc[t][r] * inv_scale;
        }
    }
}

// ---------------------------------------------------------------------------
extern "C" void kernel_launch(void* const* d_in, const int* in_sizes, int n_in,
                              void* d_out, int out_size, void* d_ws, size_t ws_size,
                              hipStream_t stream) {
    (void)in_sizes; (void)n_in; (void)out_size; (void)ws_size;
    const float* x  = (const float*)d_in[0];
    const float* mi = (const float*)d_in[1];
    const float* WQ = (const float*)d_in[2];
    const float* WK = (const float*)d_in[3];
    // d_in[4] = WV : computed by reference but unused downstream -> skipped.
    float* out = (float*)d_out;

    // Workspace: Q/K as bf16 hi/lo pairs, 4 MB each (16 MB total).
    const size_t per = (size_t)N_B * L * DK;
    __bf16* qh = (__bf16*)d_ws;
    __bf16* ql = qh + per;
    __bf16* kh = ql + per;
    __bf16* kl = kh + per;

    proj_qk_kernel<<<dim3(L / 64, N_B), 256, 0, stream>>>(x, WQ, WK, qh, ql, kh, kl);
    scores_kernel<<<dim3(L / 128, L / 128, N_B), 256, 0, stream>>>(qh, ql, kh, kl, mi, out);
}
